// Transformer_51874615001180
// MI455X (gfx1250) — matrix operand order repaired and checked by reading, not compiled, not run on hardware
//
#include <hip/hip_runtime.h>

typedef __bf16 bf16;
typedef __attribute__((ext_vector_type(8)))  __bf16 v8bf;
typedef __attribute__((ext_vector_type(16))) __bf16 v16bf;
typedef __attribute__((ext_vector_type(8)))  float  v8f;
typedef __attribute__((ext_vector_type(4)))  int    v4i;

#define AS1 __attribute__((address_space(1)))
#define AS3 __attribute__((address_space(3)))

#define S_  4096
#define D_  1024
#define H_  16
#define L_  4
#define DH_ 64
#define DF_ 4096
#define V_  50257

#define BM 128
#define BN 64
#define BK 64

static __device__ __forceinline__ v16bf mk16(v8bf lo, v8bf hi) {
  v16bf r;
#pragma unroll
  for (int i = 0; i < 8; ++i) { r[i] = lo[i]; r[i + 8] = hi[i]; }
  return r;
}

static __device__ __forceinline__ v8f wmma_bf16(v16bf a, v16bf b, v8f c) {
  // D = A(16x32 bf16) * B(32x16 bf16) + C(16x16 f32)
  return __builtin_amdgcn_wmma_f32_16x16x32_bf16(false, a, false, b, (short)0, c,
                                                 false, false);
}

// ---------------------------------------------------------------------------
// CDNA5 async global->LDS copy (16B per lane), tracked by ASYNCcnt.
// ---------------------------------------------------------------------------
static __device__ __forceinline__ void async_cp_b128(const void* g, void* l) {
#if __has_builtin(__builtin_amdgcn_global_load_async_to_lds_b128)
  __builtin_amdgcn_global_load_async_to_lds_b128((AS1 v4i*)g, (AS3 v4i*)l, 0, 0);
#else
  unsigned lofs = (unsigned)(size_t)(AS3 void*)l;
  asm volatile("global_load_async_to_lds_b128 %0, %1, off"
               :: "v"(lofs), "v"(g) : "memory");
#endif
}

template <int N>
static __device__ __forceinline__ void wait_asynccnt() {
#if __has_builtin(__builtin_amdgcn_s_wait_asynccnt)
  __builtin_amdgcn_s_wait_asynccnt(N);
#else
  asm volatile("s_wait_asynccnt %0" :: "n"(N) : "memory");
#endif
}

// ---------------------------------------------------------------------------
// Weight convert + transpose: src f32 [K,N] -> dst bf16 [N,K]  (batched, z dim)
// ---------------------------------------------------------------------------
__global__ void tcvt_kernel(const float* __restrict__ src, bf16* __restrict__ dst,
                            int K, int N) {
  size_t b  = blockIdx.z;
  size_t bs = (size_t)K * (size_t)N;
  int n = blockIdx.x * 32 + threadIdx.x;
  int k = blockIdx.y * 8 + threadIdx.y;
  if (n < N && k < K)
    dst[b * bs + (size_t)n * K + k] = (bf16)src[b * bs + (size_t)k * N + n];
}

// ---------------------------------------------------------------------------
// Embedding gather + sinusoidal positional encoding -> x (f32) and xb (bf16)
// ---------------------------------------------------------------------------
__global__ __launch_bounds__(256)
void embed_kernel(const int* __restrict__ tok, const float* __restrict__ emb,
                  float* __restrict__ x, bf16* __restrict__ xb) {
  int s = blockIdx.x;
  int t = threadIdx.x;
  int token = tok[s];
  const float LN1E4 = 9.210340371976184f;  // log(10000)
#pragma unroll
  for (int i = 0; i < 4; ++i) {
    int d = t + i * 256;
    float fi  = (d < D_ / 2) ? (float)d : (float)(d - D_ / 2);
    float ang = (float)s * __expf(-(2.0f * fi / (float)D_) * LN1E4);
    float pe  = (d < D_ / 2) ? sinf(ang) : cosf(ang);
    float v   = emb[(size_t)token * D_ + d] + pe;
    x[(size_t)s * D_ + d]  = v;
    xb[(size_t)s * D_ + d] = (bf16)v;
  }
}

// ---------------------------------------------------------------------------
// Tiled bf16 WMMA GEMM with double-buffered async global->LDS staging.
//   C[M,N] = A[M,K] @ Bt[N,K]^T + bias
//   A row-major bf16; Bt = B pre-transposed ([N][K]) so B-fragments are
//   two 16B K-runs/lane. Optional f32 out (Cf), bf16 out (Cb), ReLU,
//   transposed bf16 store (Cb[n*M+m]). Batched over blockIdx.z.
//   Requires M % 128 == 0, K % 64 == 0; Bt rows must be readable up to the
//   next multiple of 64 (caller pads allocation; epilogue guards col < N).
// ---------------------------------------------------------------------------
__global__ __launch_bounds__(256)
void gemm_bt_kernel(const bf16* __restrict__ A, const bf16* __restrict__ Bt,
                    const float* __restrict__ bias,
                    float* __restrict__ Cf, bf16* __restrict__ Cb,
                    int M, int N, int K,
                    size_t sA, size_t sB, size_t sBias, size_t sC,
                    int relu, int transC) {
  __shared__ __align__(16) bf16 As[2][BM * BK];  // 2 x 16 KB
  __shared__ __align__(16) bf16 Bs[2][BN * BK];  // 2 x  8 KB

  const int tid  = threadIdx.x;
  const int wave = tid >> 5;
  const int lane = tid & 31;
  const int lrow = lane & 15;
  const int halo = (lane < 16) ? 0 : 1;
  const int n0 = blockIdx.x * BN;
  const int m0 = blockIdx.y * BM;
  const size_t b = blockIdx.z;
  A  += b * sA;
  Bt += b * sB;

  v8f acc[4];
#pragma unroll
  for (int t = 0; t < 4; ++t)
#pragma unroll
    for (int v = 0; v < 8; ++v) acc[t][v] = 0.0f;

  // Issue one stage of async copies: 4 A-chunks + 2 B-chunks per thread.
  auto stage = [&](int buf, int k0) {
#pragma unroll
    for (int i = 0; i < 4; ++i) {
      int chunk = tid + i * 256;
      int r = chunk >> 3, c = (chunk & 7) * 8;
      async_cp_b128(&A[(size_t)(m0 + r) * K + k0 + c], &As[buf][r * BK + c]);
    }
#pragma unroll
    for (int i = 0; i < 2; ++i) {
      int chunk = tid + i * 256;
      int r = chunk >> 3, c = (chunk & 7) * 8;
      async_cp_b128(&Bt[(size_t)(n0 + r) * K + k0 + c], &Bs[buf][r * BK + c]);
    }
  };

  const int nStage = K / BK;
  stage(0, 0);
  int buf = 0;
  for (int it = 0; it < nStage; ++it) {
    if (it + 1 < nStage) {
      stage(buf ^ 1, (it + 1) * BK);
      wait_asynccnt<6>();   // in-order: <=6 outstanding => current buf landed
    } else {
      wait_asynccnt<0>();
    }
    __syncthreads();

#pragma unroll
    for (int kk = 0; kk < BK; kk += 32) {
      // Load ALL fragments for this kk first, then run WMMAs back-to-back so
      // the scheduler needs only one dscnt wait per step.
      const bf16* ar = &As[buf][(wave * 16 + lrow) * BK + kk + halo * 8];
      v16bf af = mk16(*(const v8bf*)ar, *(const v8bf*)(ar + 16));
      v16bf bfr[4];
#pragma unroll
      for (int nt = 0; nt < 4; ++nt) {
        const bf16* br = &Bs[buf][(nt * 16 + lrow) * BK + kk + halo * 8];
        bfr[nt] = mk16(*(const v8bf*)br, *(const v8bf*)(br + 16));
      }
#pragma unroll
      for (int nt = 0; nt < 4; ++nt)
        acc[nt] = wmma_bf16(af, bfr[nt], acc[nt]);
    }
    __syncthreads();
    buf ^= 1;
  }

  // Epilogue: C layout -> VGPR v holds rows v / v+8 across lane halves
#pragma unroll
  for (int nt = 0; nt < 4; ++nt) {
    int col = n0 + nt * 16 + lrow;
    if (col < N) {
      float bv = bias ? bias[b * sBias + col] : 0.0f;
#pragma unroll
      for (int v = 0; v < 8; ++v) {
        int row = m0 + wave * 16 + v + halo * 8;
        float val = acc[nt][v] + bv;
        if (relu) val = fmaxf(val, 0.0f);
        if (Cf) Cf[b * sC + (size_t)row * N + col] = val;
        if (Cb) {
          if (transC) Cb[b * sC + (size_t)col * M + row] = (bf16)val;
          else        Cb[b * sC + (size_t)row * N + col] = (bf16)val;
        }
      }
    }
  }
}

// ---------------------------------------------------------------------------
// Flash attention: per wave one 16-row query tile, 32-key blocks.
// q,k: [H][S][DH] bf16; vT: [H][DH][S] bf16; y: [S][D] f32 (heads concat).
// QK^T and P*V on WMMA; online softmax via intra-half shuffles; P converted
// from C-layout to A-layout through a per-wave LDS tile.
// ---------------------------------------------------------------------------
__global__ __launch_bounds__(256)
void attn_kernel(const bf16* __restrict__ q, const bf16* __restrict__ k,
                 const bf16* __restrict__ vT, float* __restrict__ y) {
  __shared__ __align__(16) bf16 Pst[8][16][32];  // 8 KB, per-wave scratch

  const int tid  = threadIdx.x;
  const int wave = tid >> 5;
  const int lane = tid & 31;
  const int lrow = lane & 15;
  const int halo = (lane < 16) ? 0 : 1;
  const int h    = blockIdx.y;
  const int row0 = blockIdx.x * 128 + wave * 16;

  const bf16* qh = q  + (size_t)h * S_ * DH_;
  const bf16* kh = k  + (size_t)h * S_ * DH_;
  const bf16* vh = vT + (size_t)h * DH_ * S_;

  v16bf qa[2];
#pragma unroll
  for (int kb = 0; kb < 2; ++kb) {
    const bf16* p = &qh[(size_t)(row0 + lrow) * DH_ + kb * 32 + halo * 8];
    qa[kb] = mk16(*(const v8bf*)p, *(const v8bf*)(p + 16));
  }

  float m[8], l[8];
  v8f acc[4];
#pragma unroll
  for (int v = 0; v < 8; ++v) { m[v] = -1e30f; l[v] = 0.0f; }
#pragma unroll
  for (int t = 0; t < 4; ++t)
#pragma unroll
    for (int v = 0; v < 8; ++v) acc[t][v] = 0.0f;

  for (int j0 = 0; j0 < S_; j0 += 32) {
    // Prefetch next key block's K and V lines into cache (global_prefetch_b8)
    if (j0 + 32 < S_) {
      __builtin_prefetch(&kh[(size_t)(j0 + 32 + lrow) * DH_], 0, 1);
      __builtin_prefetch(&vh[(size_t)lrow * S_ + j0 + 32], 0, 1);
    }

    // QK^T: load all 4 K-fragments, then 4 WMMAs
    v16bf kf[2][2];
#pragma unroll
    for (int nt = 0; nt < 2; ++nt)
#pragma unroll
      for (int kb = 0; kb < 2; ++kb) {
        const bf16* p =
            &kh[(size_t)(j0 + nt * 16 + lrow) * DH_ + kb * 32 + halo * 8];
        kf[nt][kb] = mk16(*(const v8bf*)p, *(const v8bf*)(p + 16));
      }
    v8f s[2];
#pragma unroll
    for (int nt = 0; nt < 2; ++nt) {
#pragma unroll
      for (int v = 0; v < 8; ++v) s[nt][v] = 0.0f;
      s[nt] = wmma_bf16(qa[0], kf[nt][0], s[nt]);
      s[nt] = wmma_bf16(qa[1], kf[nt][1], s[nt]);
    }

    // Online softmax (scale = 1/sqrt(64) = 0.125)
#pragma unroll
    for (int v = 0; v < 8; ++v) {
      float a0 = s[0][v] * 0.125f, a1 = s[1][v] * 0.125f;
      float tm = fmaxf(a0, a1);
#pragma unroll
      for (int msk = 1; msk < 16; msk <<= 1)
        tm = fmaxf(tm, __shfl_xor(tm, msk, 32));
      float mn    = fmaxf(m[v], tm);
      float alpha = __expf(m[v] - mn);
      m[v] = mn;
      float e0 = __expf(a0 - mn), e1 = __expf(a1 - mn);
      s[0][v] = e0; s[1][v] = e1;
      float rs = e0 + e1;
#pragma unroll
      for (int msk = 1; msk < 16; msk <<= 1) rs += __shfl_xor(rs, msk, 32);
      l[v] = l[v] * alpha + rs;
#pragma unroll
      for (int t = 0; t < 4; ++t) acc[t][v] *= alpha;
    }

    // P: C-layout -> LDS (row-major 16x32) -> A-layout fragment
#pragma unroll
    for (int v = 0; v < 8; ++v) {
      int r = v + halo * 8;
      Pst[wave][r][lrow]      = (bf16)s[0][v];
      Pst[wave][r][16 + lrow] = (bf16)s[1][v];
    }
    const bf16* pp = &Pst[wave][lrow][halo * 8];
    v16bf pf = mk16(*(const v8bf*)pp, *(const v8bf*)(pp + 16));

    // P(16x32) @ V(32x64): load all 4 V-fragments, then 4 WMMAs
    v16bf vf[4];
#pragma unroll
    for (int t = 0; t < 4; ++t) {
      const bf16* vp = &vh[(size_t)(t * 16 + lrow) * S_ + j0 + halo * 8];
      vf[t] = mk16(*(const v8bf*)vp, *(const v8bf*)(vp + 16));
    }
#pragma unroll
    for (int t = 0; t < 4; ++t)
      acc[t] = wmma_bf16(pf, vf[t], acc[t]);
  }

#pragma unroll
  for (int t = 0; t < 4; ++t) {
    int col = h * DH_ + t * 16 + lrow;
#pragma unroll
    for (int v = 0; v < 8; ++v) {
      int row = row0 + v + halo * 8;
      y[(size_t)row * D_ + col] = acc[t][v] / l[v];
    }
  }
}

// ---------------------------------------------------------------------------
// x = LayerNorm(x + y) * g + b; writes f32 x and bf16 xb. One block per row.
// ---------------------------------------------------------------------------
__global__ __launch_bounds__(256)
void add_ln_kernel(const float* __restrict__ x, const float* __restrict__ yv,
                   const float* __restrict__ g, const float* __restrict__ bta,
                   float* __restrict__ xo, bf16* __restrict__ xob) {
  __shared__ float red[256];
  int row = blockIdx.x, t = threadIdx.x;
  float v[4];
  float sum = 0.0f;
#pragma unroll
  for (int i = 0; i < 4; ++i) {
    int d = t + i * 256;
    v[i] = x[(size_t)row * D_ + d] + yv[(size_t)row * D_ + d];
    sum += v[i];
  }
  red[t] = sum;
  __syncthreads();
  for (int o = 128; o > 0; o >>= 1) {
    if (t < o) red[t] += red[t + o];
    __syncthreads();
  }
  float mean = red[0] / (float)D_;
  __syncthreads();
  float vs = 0.0f;
#pragma unroll
  for (int i = 0; i < 4; ++i) { float d = v[i] - mean; vs += d * d; }
  red[t] = vs;
  __syncthreads();
  for (int o = 128; o > 0; o >>= 1) {
    if (t < o) red[t] += red[t + o];
    __syncthreads();
  }
  float rstd = rsqrtf(red[0] / (float)D_ + 1e-5f);
#pragma unroll
  for (int i = 0; i < 4; ++i) {
    int d = t + i * 256;
    float o = (v[i] - mean) * rstd * g[d] + bta[d];
    xo[(size_t)row * D_ + d]  = o;
    xob[(size_t)row * D_ + d] = (bf16)o;
  }
}

// ---------------------------------------------------------------------------
// Row softmax over V=50257, in place over logits. One block per row.
// ---------------------------------------------------------------------------
__global__ __launch_bounds__(256)
void softmax_kernel(float* __restrict__ p) {
  __shared__ float red[256];
  int row = blockIdx.x, t = threadIdx.x;
  float* pr = p + (size_t)row * V_;
  float mx = -1e30f;
  for (int j = t; j < V_; j += 256) mx = fmaxf(mx, pr[j]);
  red[t] = mx;
  __syncthreads();
  for (int o = 128; o > 0; o >>= 1) {
    if (t < o) red[t] = fmaxf(red[t], red[t + o]);
    __syncthreads();
  }
  mx = red[0];
  __syncthreads();
  float sum = 0.0f;
  for (int j = t; j < V_; j += 256) {
    float e = __expf(pr[j] - mx);
    pr[j] = e;
    sum += e;
  }
  red[t] = sum;
  __syncthreads();
  for (int o = 128; o > 0; o >>= 1) {
    if (t < o) red[t] += red[t + o];
    __syncthreads();
  }
  float inv = 1.0f / red[0];
  for (int j = t; j < V_; j += 256) pr[j] *= inv;
}

// ---------------------------------------------------------------------------
extern "C" void kernel_launch(void* const* d_in, const int* in_sizes, int n_in,
                              void* d_out, int out_size, void* d_ws, size_t ws_size,
                              hipStream_t stream) {
  (void)in_sizes; (void)n_in; (void)out_size; (void)ws_size;
  const int*   tokens = (const int*)  d_in[0];
  const float* emb    = (const float*)d_in[1];
  const float* Wq     = (const float*)d_in[2];
  const float* bq     = (const float*)d_in[3];
  const float* Wk     = (const float*)d_in[4];
  const float* bk     = (const float*)d_in[5];
  const float* Wv     = (const float*)d_in[6];
  const float* bv     = (const float*)d_in[7];
  const float* ln1g   = (const float*)d_in[8];
  const float* ln1b   = (const float*)d_in[9];
  const float* W1     = (const float*)d_in[10];
  const float* b1     = (const float*)d_in[11];
  const float* W2     = (const float*)d_in[12];
  const float* b2     = (const float*)d_in[13];
  const float* ln2g   = (const float*)d_in[14];
  const float* ln2b   = (const float*)d_in[15];
  const float* Wout   = (const float*)d_in[16];
  const float* bout   = (const float*)d_in[17];
  float* out = (float*)d_out;

  char* ws = (char*)d_ws;
  size_t off = 0;
  auto alloc = [&](size_t bytes) -> void* {
    void* p = ws + off;
    off = (off + bytes + 255) & ~(size_t)255;
    return p;
  };
  float* x    = (float*)alloc((size_t)S_ * D_ * 4);
  bf16*  xb   = (bf16*) alloc((size_t)S_ * D_ * 2);
  float* yv   = (float*)alloc((size_t)S_ * D_ * 4);
  bf16*  qb   = (bf16*) alloc((size_t)S_ * D_ * 2);
  bf16*  kb_  = (bf16*) alloc((size_t)S_ * D_ * 2);
  bf16*  vTb  = (bf16*) alloc((size_t)S_ * D_ * 2);
  bf16*  hb   = (bf16*) alloc((size_t)S_ * DF_ * 2);
  bf16*  Wqb  = (bf16*) alloc((size_t)L_ * H_ * DH_ * D_ * 2);
  bf16*  Wkb  = (bf16*) alloc((size_t)L_ * H_ * DH_ * D_ * 2);
  bf16*  Wvb  = (bf16*) alloc((size_t)L_ * H_ * DH_ * D_ * 2);
  bf16*  W1b  = (bf16*) alloc((size_t)L_ * DF_ * D_ * 2);
  bf16*  W2b  = (bf16*) alloc((size_t)L_ * D_ * DF_ * 2);
  // Pad Wout^T by 64 rows so ragged-N async staging never reads OOB.
  bf16*  Wob  = (bf16*) alloc((size_t)(V_ + BN) * D_ * 2);

  // Weight convert + transpose to bf16 [N][K]
  dim3 tb(32, 8);
  tcvt_kernel<<<dim3((DH_ + 31) / 32, (D_ + 7) / 8, L_ * H_), tb, 0, stream>>>(Wq, Wqb, D_, DH_);
  tcvt_kernel<<<dim3((DH_ + 31) / 32, (D_ + 7) / 8, L_ * H_), tb, 0, stream>>>(Wk, Wkb, D_, DH_);
  tcvt_kernel<<<dim3((DH_ + 31) / 32, (D_ + 7) / 8, L_ * H_), tb, 0, stream>>>(Wv, Wvb, D_, DH_);
  tcvt_kernel<<<dim3((DF_ + 31) / 32, (D_ + 7) / 8, L_), tb, 0, stream>>>(W1, W1b, D_, DF_);
  tcvt_kernel<<<dim3((D_ + 31) / 32, (DF_ + 7) / 8, L_), tb, 0, stream>>>(W2, W2b, DF_, D_);
  tcvt_kernel<<<dim3((V_ + 31) / 32, (D_ + 7) / 8, 1), tb, 0, stream>>>(Wout, Wob, D_, V_);

  embed_kernel<<<S_, 256, 0, stream>>>(tokens, emb, x, xb);

  for (int l = 0; l < L_; ++l) {
    const bf16* wq = Wqb + (size_t)l * H_ * DH_ * D_;
    const bf16* wk = Wkb + (size_t)l * H_ * DH_ * D_;
    const bf16* wv = Wvb + (size_t)l * H_ * DH_ * D_;

    // Q/K: [H][S][DH] bf16; V stored transposed [H][DH][S] for the PV WMMA
    gemm_bt_kernel<<<dim3(1, S_ / BM, H_), 256, 0, stream>>>(
        xb, wq, bq + (size_t)l * H_ * DH_, nullptr, qb,
        S_, DH_, D_, 0, (size_t)DH_ * D_, DH_, (size_t)S_ * DH_, 0, 0);
    gemm_bt_kernel<<<dim3(1, S_ / BM, H_), 256, 0, stream>>>(
        xb, wk, bk + (size_t)l * H_ * DH_, nullptr, kb_,
        S_, DH_, D_, 0, (size_t)DH_ * D_, DH_, (size_t)S_ * DH_, 0, 0);
    gemm_bt_kernel<<<dim3(1, S_ / BM, H_), 256, 0, stream>>>(
        xb, wv, bv + (size_t)l * H_ * DH_, nullptr, vTb,
        S_, DH_, D_, 0, (size_t)DH_ * D_, DH_, (size_t)DH_ * S_, 0, 1);

    attn_kernel<<<dim3(S_ / 128, H_), 256, 0, stream>>>(qb, kb_, vTb, yv);

    add_ln_kernel<<<S_, 256, 0, stream>>>(x, yv, ln1g + (size_t)l * D_,
                                          ln1b + (size_t)l * D_, x, xb);

    // FFN: h = relu(x@W1+b1) -> bf16; y = h@W2+b2 -> f32
    gemm_bt_kernel<<<dim3(DF_ / BN, S_ / BM, 1), 256, 0, stream>>>(
        xb, W1b + (size_t)l * DF_ * D_, b1 + (size_t)l * DF_, nullptr, hb,
        S_, DF_, D_, 0, 0, 0, 0, 1, 0);
    gemm_bt_kernel<<<dim3(D_ / BN, S_ / BM, 1), 256, 0, stream>>>(
        hb, W2b + (size_t)l * D_ * DF_, b2 + (size_t)l * D_, yv, nullptr,
        S_, D_, DF_, 0, 0, 0, 0, 0, 0);

    add_ln_kernel<<<S_, 256, 0, stream>>>(x, yv, ln2g + (size_t)l * D_,
                                          ln2b + (size_t)l * D_, x, xb);
  }

  // logits = x @ Wout + bout -> d_out, then row softmax in place
  gemm_bt_kernel<<<dim3((V_ + BN - 1) / BN, S_ / BM, 1), 256, 0, stream>>>(
      xb, Wob, bout, out, nullptr, S_, V_, D_, 0, 0, 0, 0, 0, 0);
  softmax_kernel<<<S_, 256, 0, stream>>>(out);
}
